// DilatedAttention_31997506355936
// MI455X (gfx1250) — matrix operand order repaired and checked by reading, not compiled, not run on hardware
//
#include <hip/hip_runtime.h>
#include <hip/hip_bf16.h>

// CDNA5 / gfx1250 implementation of the reference "dilated attention".
// Per position: tiny g x g attention over a head group (g = 6/5/5, d = 64),
// padded to 16x16 and computed with v_wmma_f32_16x16x32_f16 (f16 in, f32 acc).
// Two passes per group: (1) accumulate per-(b,head,d) sums via f32 atomics,
// (2) recompute x and write out += x / (3 * sum). Memory-bound by design.

typedef __attribute__((ext_vector_type(16))) _Float16 v16h;
typedef __attribute__((ext_vector_type(8)))  float    v8f;

#define WAVES_PER_BLOCK 8
#define POS_PER_WAVE    16
#define SEQ_LEN         8192
#define NUM_HEADS       16
#define HEAD_DIM        64

__device__ __forceinline__ float halfwave_max(float v) {
#pragma unroll
  for (int m = 1; m <= 8; m <<= 1) v = fmaxf(v, __shfl_xor(v, m, 32));
  return v;
}
__device__ __forceinline__ float halfwave_sum(float v) {
#pragma unroll
  for (int m = 1; m <= 8; m <<= 1) v += __shfl_xor(v, m, 32);
  return v;
}

// G: heads in group, R: dilation rate, SEG: segment length, OFF: dilation offset,
// HMIN: first head, NSEG: segments per batch, PASS2: write-out pass.
template <int G, int R, int SEG, int OFF, int HMIN, int NSEG, bool PASS2>
__global__ __launch_bounds__(256) void dil_group_kernel(
    const float* __restrict__ Qg, const float* __restrict__ Kg,
    const float* __restrict__ Vg, float* __restrict__ sums,
    float* __restrict__ out) {
  constexpr int SPRIME        = SEG / R;          // 2048 for every group
  constexpr int POS_PER_BATCH = NSEG * SPRIME;    // multiple of POS_PER_WAVE

  const int lane  = threadIdx.x & 31;
  const int wv    = threadIdx.x >> 5;
  const int idx16 = lane & 15;   // row (A/C) or column (B/C) index
  const int hl    = lane >> 4;   // which 16-lane half of the wave

  // +1 pad avoids bank conflicts on the transposed read.
  __shared__ float lds_attn[WAVES_PER_BLOCK][16][17];

  const int waveGlobal = blockIdx.x * WAVES_PER_BLOCK + wv;
  const int posBase    = waveGlobal * POS_PER_WAVE;
  const int bIdx       = posBase / POS_PER_BATCH;  // whole chunk shares batch

  float acc[4][8];
#pragma unroll
  for (int cd = 0; cd < 4; ++cd)
#pragma unroll
    for (int j = 0; j < 8; ++j) acc[cd][j] = 0.0f;

  float inv3s[4][8];
  if (PASS2) {
#pragma unroll
    for (int cd = 0; cd < 4; ++cd)
#pragma unroll
      for (int j = 0; j < 8; ++j) {
        float sv = 1.0f;
        if (j < G) sv = sums[(bIdx * G + j) * HEAD_DIM + cd * 16 + idx16];
        inv3s[cd][j] = 1.0f / (3.0f * sv);
      }
  }

#pragma unroll 1
  for (int pp = 0; pp < POS_PER_WAVE; ++pp) {
    const int posFlat = posBase + pp;
    const int rem     = posFlat - bIdx * POS_PER_BATCH;
    const int nIdx    = rem / SPRIME;
    const int jj      = rem - nIdx * SPRIME;
    const int p       = nIdx * SEG + OFF + jj * R;

    const size_t rowBase =
        (((size_t)bIdx * SEQ_LEN + p) * NUM_HEADS + HMIN) * HEAD_DIM;
    const float* qb = Qg + rowBase;
    const float* kb = Kg + rowBase;
    const float* vb = Vg + rowBase;

    // Prefetch next position's Q/K rows (global_prefetch_b8).
    if (pp + 1 < POS_PER_WAVE) {
      const int rem2  = rem + 1;
      const int nIdx2 = rem2 / SPRIME;
      const int p2    = nIdx2 * SEG + OFF + (rem2 - nIdx2 * SPRIME) * R;
      const size_t rb2 =
          (((size_t)bIdx * SEQ_LEN + p2) * NUM_HEADS + HMIN) * HEAD_DIM;
      __builtin_prefetch(Qg + rb2 + (size_t)idx16 * HEAD_DIM, 0, 1);
      __builtin_prefetch(Kg + rb2 + (size_t)idx16 * HEAD_DIM, 0, 1);
    }

    // ---- Build A (Q, 16x32 f16 per K-chunk) and B (K^T, 32x16 f16) ----
    // A layout: lanes<16 elems 0..7 -> K c*32+0..7, elems 8..15 -> K c*32+16..23;
    //           lanes>=16: +8 on the first run, +24 on the second.
    // B layout: lane holds column n=lane%16; same K order as A:
    //           elems 0..7 -> K c*32+(lane/16)*8+0..7, elems 8..15 -> +16.
    v16h Aq[2], Bk[2];
#pragma unroll
    for (int c = 0; c < 2; ++c) {
      v16h a, bm;
#pragma unroll
      for (int t = 0; t < 16; ++t) { a[t] = (_Float16)0.0f; bm[t] = (_Float16)0.0f; }
      if (idx16 < G) {
        const float* r = qb + idx16 * HEAD_DIM + c * 32 + hl * 8;
        float4 f0 = *(const float4*)(r);
        float4 f1 = *(const float4*)(r + 4);
        float4 f2 = *(const float4*)(r + 16);
        float4 f3 = *(const float4*)(r + 20);
        a[0]=(_Float16)f0.x; a[1]=(_Float16)f0.y; a[2]=(_Float16)f0.z; a[3]=(_Float16)f0.w;
        a[4]=(_Float16)f1.x; a[5]=(_Float16)f1.y; a[6]=(_Float16)f1.z; a[7]=(_Float16)f1.w;
        a[8]=(_Float16)f2.x; a[9]=(_Float16)f2.y; a[10]=(_Float16)f2.z; a[11]=(_Float16)f2.w;
        a[12]=(_Float16)f3.x; a[13]=(_Float16)f3.y; a[14]=(_Float16)f3.z; a[15]=(_Float16)f3.w;

        const float* rk = kb + idx16 * HEAD_DIM + c * 32 + hl * 8;
        float4 g0 = *(const float4*)(rk);
        float4 g1 = *(const float4*)(rk + 4);
        float4 g2 = *(const float4*)(rk + 16);
        float4 g3 = *(const float4*)(rk + 20);
        bm[0]=(_Float16)g0.x; bm[1]=(_Float16)g0.y; bm[2]=(_Float16)g0.z; bm[3]=(_Float16)g0.w;
        bm[4]=(_Float16)g1.x; bm[5]=(_Float16)g1.y; bm[6]=(_Float16)g1.z; bm[7]=(_Float16)g1.w;
        bm[8]=(_Float16)g2.x; bm[9]=(_Float16)g2.y; bm[10]=(_Float16)g2.z; bm[11]=(_Float16)g2.w;
        bm[12]=(_Float16)g3.x; bm[13]=(_Float16)g3.y; bm[14]=(_Float16)g3.z; bm[15]=(_Float16)g3.w;
      }
      Aq[c] = a;
      Bk[c] = bm;
    }

    // ---- Scores: S[16x16] = Q * K^T  (two K=32 chunks over d=64) ----
    v8f S = {0.f, 0.f, 0.f, 0.f, 0.f, 0.f, 0.f, 0.f};
    S = __builtin_amdgcn_wmma_f32_16x16x32_f16(false, Aq[0], false, Bk[0],
                                               (short)0, S, false, false);
    S = __builtin_amdgcn_wmma_f32_16x16x32_f16(false, Aq[1], false, Bk[1],
                                               (short)0, S, false, false);

    // ---- Softmax over the key-head axis (columns), rows in C/D layout ----
#pragma unroll
    for (int j = 0; j < 8; ++j) {
      float sc = (idx16 < G) ? (S[j] * 0.125f) : -3.0e38f;
      float mx = halfwave_max(sc);
      float e  = (idx16 < G) ? __expf(sc - mx) : 0.0f;
      float sm = halfwave_sum(e);
      lds_attn[wv][j + hl * 8][idx16] = e / sm;  // padded cols are exactly 0
    }
    __syncthreads();

    // ---- Transpose attn into A-operand layout (K padded 16 -> 32) ----
    v16h Aat;
#pragma unroll
    for (int t = 0; t < 8; ++t) {
      Aat[t]     = (_Float16)lds_attn[wv][idx16][hl * 8 + t];
      Aat[t + 8] = (_Float16)0.0f;
    }
    __syncthreads();

    // ---- x = attn * V : four N=16 chunks over d=64, K padded to 32 ----
#pragma unroll
    for (int cd = 0; cd < 4; ++cd) {
      v16h Bv;
#pragma unroll
      for (int t = 0; t < 16; ++t) {
        float vvv = 0.0f;
        if (hl == 0 && t < G) vvv = vb[t * HEAD_DIM + cd * 16 + idx16];
        Bv[t] = (_Float16)vvv;
      }
      v8f X = {0.f, 0.f, 0.f, 0.f, 0.f, 0.f, 0.f, 0.f};
      X = __builtin_amdgcn_wmma_f32_16x16x32_f16(false, Aat, false, Bv,
                                                 (short)0, X, false, false);
      if (!PASS2) {
#pragma unroll
        for (int j = 0; j < 8; ++j) acc[cd][j] += X[j];
      } else if (hl == 0) {
#pragma unroll
        for (int j = 0; j < G; ++j) {
          size_t oidx =
              (((size_t)bIdx * SEQ_LEN + p) * NUM_HEADS + HMIN + j) * HEAD_DIM +
              cd * 16 + idx16;
          out[oidx] += X[j] * inv3s[cd][j];
        }
      }
    }
  }

  if (!PASS2) {
    if (hl == 0) {
#pragma unroll
      for (int cd = 0; cd < 4; ++cd)
#pragma unroll
        for (int j = 0; j < G; ++j)
          atomicAdd(&sums[(bIdx * G + j) * HEAD_DIM + cd * 16 + idx16],
                    acc[cd][j]);
    }
  }
}

__global__ void zero_f32_kernel(float* __restrict__ p, size_t n) {
  size_t i      = (size_t)blockIdx.x * blockDim.x + threadIdx.x;
  size_t stride = (size_t)gridDim.x * blockDim.x;
  for (; i < n; i += stride) p[i] = 0.0f;
}

extern "C" void kernel_launch(void* const* d_in, const int* in_sizes, int n_in,
                              void* d_out, int out_size, void* d_ws,
                              size_t ws_size, hipStream_t stream) {
  (void)in_sizes; (void)n_in; (void)ws_size;
  const float* Q = (const float*)d_in[0];
  const float* K = (const float*)d_in[1];
  const float* V = (const float*)d_in[2];
  float* out  = (float*)d_out;
  float* sums = (float*)d_ws;  // group0 @0 (4*6*64), group1 @1536, group2 @2816

  // Zero output and the (poisoned) sum scratch.
  zero_f32_kernel<<<2048, 256, 0, stream>>>(out, (size_t)out_size);
  zero_f32_kernel<<<16, 256, 0, stream>>>(sums, (size_t)4096);

  // Pass 1: accumulate per-(b, head, d) sums of x.
  // blocks = positions / (POS_PER_WAVE * WAVES_PER_BLOCK) — exact divisions.
  dil_group_kernel<6, 1, 2048, 0, 0, 4, false>
      <<<256, 256, 0, stream>>>(Q, K, V, sums + 0, out);
  dil_group_kernel<5, 2, 4096, 1, 5, 2, false>
      <<<128, 256, 0, stream>>>(Q, K, V, sums + 1536, out);
  dil_group_kernel<5, 4, 8192, 2, 10, 1, false>
      <<<64, 256, 0, stream>>>(Q, K, V, sums + 2816, out);

  // Pass 2: recompute x, out += x / (3 * sum). Stream order serializes the
  // head-5 overlap between groups 0 and 1.
  dil_group_kernel<6, 1, 2048, 0, 0, 4, true>
      <<<256, 256, 0, stream>>>(Q, K, V, sums + 0, out);
  dil_group_kernel<5, 2, 4096, 1, 5, 2, true>
      <<<128, 256, 0, stream>>>(Q, K, V, sums + 1536, out);
  dil_group_kernel<5, 4, 8192, 2, 10, 1, true>
      <<<64, 256, 0, stream>>>(Q, K, V, sums + 2816, out);
}
